// GATv2GraphClassifier_52956946760188
// MI455X (gfx1250) — matrix operand order repaired and checked by reading, not compiled, not run on hardware
//
#include <hip/hip_runtime.h>
#include <hip/hip_bf16.h>
#include <math.h>

#define N_NODES_C   100000
#define N_EDGES_C   1600000
#define TE_EDGES_C  (N_EDGES_C + N_NODES_C)
#define HID_C       64
#define NGRAPH_C    512
#define NEG_SLOPE_C 0.2f

typedef __attribute__((ext_vector_type(16))) __bf16 v16bf;
typedef __attribute__((ext_vector_type(8)))  float  v8f;

union FragBF { v16bf v; uint4 q[2]; };

__device__ __forceinline__ unsigned short f2bf(float f) {
  unsigned u = __float_as_uint(f);
  u += 0x7FFFu + ((u >> 16) & 1u);           // round-to-nearest-even
  return (unsigned short)(u >> 16);
}

// float atomic max via integer atomics (monotone bit tricks); init must be -inf
__device__ __forceinline__ void atomicMaxF(float* addr, float v) {
  if (v >= 0.0f) atomicMax((int*)addr, __float_as_int(v));
  else           atomicMin((unsigned int*)addr, __float_as_uint(v));
}

// ---------------- conversion kernels ----------------
__global__ void convX_kernel(const float* __restrict__ x, unsigned short* __restrict__ xb, size_t n) {
  size_t i = (size_t)blockIdx.x * blockDim.x + threadIdx.x;
  if (i < n) xb[i] = f2bf(x[i]);
}

// W [K,64] row-major f32 -> Wt [64,K] col-major bf16
__global__ void convW_kernel(const float* __restrict__ W, unsigned short* __restrict__ Wt, int K) {
  int idx = blockIdx.x * blockDim.x + threadIdx.x;
  if (idx >= K * 64) return;
  int k = idx >> 6, c = idx & 63;
  Wt[c * K + k] = f2bf(W[idx]);
}

__global__ void fill_kernel(float* __restrict__ p, float v, size_t n) {
  size_t i = (size_t)blockIdx.x * blockDim.x + threadIdx.x;
  if (i < n) p[i] = v;
}

// ---------------- dual WMMA GEMM: xl = Xb@Wl + bl, xr = Xb@Wr + br ----------------
// Xb: [nrows, K] bf16 row-major. Wl/Wr: [64, K] bf16 col-major. Outputs [nrows, 64] f32.
template <int K>
__global__ __launch_bounds__(256) void gemm_dual_kernel(
    const unsigned short* __restrict__ Xb,
    const unsigned short* __restrict__ Wl,
    const unsigned short* __restrict__ Wr,
    const float* __restrict__ bl, const float* __restrict__ br,
    float* __restrict__ xl, float* __restrict__ xr, int nrows) {
  const int lane = threadIdx.x & 31;
  const int wave = threadIdx.x >> 5;          // 8 waves: 2 row tiles x 4 col tiles
  const int rt = wave >> 2, ct = wave & 3;
  const int half = lane >> 4;                 // 0 or 1
  const int rbase = blockIdx.x * 32 + rt * 16;
  int row = rbase + (lane & 15);
  int rowc = row < nrows ? row : (nrows - 1); // clamp loads, guard stores
  const int col = ct * 16 + (lane & 15);

  v8f accl = {}; v8f accr = {};
#pragma unroll
  for (int kt = 0; kt < K / 32; ++kt) {
    FragBF a, fl, fr;
    // A 16x32 bf16: lane half selects K=kt*32 or +16, 16 contiguous
    const int ka = kt * 32 + half * 16;
    const unsigned short* ap = Xb + (size_t)rowc * K + ka;
    a.q[0] = *(const uint4*)(ap);
    a.q[1] = *(const uint4*)(ap + 8);
    // B 32x16 bf16 col-major: lane half selects K=kt*32 or +16, 16 contiguous
    const int kb = kt * 32 + half * 16;
    const unsigned short* lp = Wl + (size_t)col * K + kb;
    const unsigned short* rp = Wr + (size_t)col * K + kb;
    fl.q[0] = *(const uint4*)(lp); fl.q[1] = *(const uint4*)(lp + 8);
    fr.q[0] = *(const uint4*)(rp); fr.q[1] = *(const uint4*)(rp + 8);
    accl = __builtin_amdgcn_wmma_f32_16x16x32_bf16(false, a.v, false, fl.v, (short)0, accl, false, false);
    accr = __builtin_amdgcn_wmma_f32_16x16x32_bf16(false, a.v, false, fr.v, (short)0, accr, false, false);
  }
  const float bla = bl[col], bra = br[col];
#pragma unroll
  for (int r = 0; r < 8; ++r) {
    int m = rbase + r + half * 8;             // D layout: M = vgpr + 8*(lane>=16)
    if (m < nrows) {
      xl[(size_t)m * 64 + col] = accl[r] + bla;
      xr[(size_t)m * 64 + col] = accr[r] + bra;
    }
  }
}

// ---------------- edge pass 1: e = att . leakyrelu(xl[s]+xr[d]); m[d] = max ----------------
__global__ void edge_pass1_kernel(const long long* __restrict__ ei,
                                  const float* __restrict__ xl, const float* __restrict__ xr,
                                  const float* __restrict__ att,
                                  float* __restrict__ ebuf, float* __restrict__ m) {
  int t = blockIdx.x * blockDim.x + threadIdx.x;
  int eidx = t >> 4;            // 16 lanes per edge, 4 channels per lane
  int sub = t & 15;
  if (eidx >= TE_EDGES_C) return;
  int s, d;
  if (eidx < N_EDGES_C) { s = (int)ei[eidx]; d = (int)ei[(size_t)N_EDGES_C + eidx]; }
  else { s = d = eidx - N_EDGES_C; }
  float4 a4 = *(const float4*)(att + sub * 4);
  float4 l4 = *(const float4*)(xl + (size_t)s * 64 + sub * 4);
  float4 r4 = *(const float4*)(xr + (size_t)d * 64 + sub * 4);
  float h, p = 0.0f;
  h = l4.x + r4.x; h = h >= 0.0f ? h : NEG_SLOPE_C * h; p += a4.x * h;
  h = l4.y + r4.y; h = h >= 0.0f ? h : NEG_SLOPE_C * h; p += a4.y * h;
  h = l4.z + r4.z; h = h >= 0.0f ? h : NEG_SLOPE_C * h; p += a4.z * h;
  h = l4.w + r4.w; h = h >= 0.0f ? h : NEG_SLOPE_C * h; p += a4.w * h;
#pragma unroll
  for (int off = 8; off >= 1; off >>= 1) p += __shfl_xor(p, off, 32);
  if (sub == 0) { ebuf[eidx] = p; atomicMaxF(m + d, p); }
}

// ---------------- edge pass 2: a=exp(e-m[d]); z[d]+=a; out[d]+=a*xl[s] ----------------
__global__ void edge_pass2_kernel(const long long* __restrict__ ei,
                                  const float* __restrict__ xl,
                                  const float* __restrict__ ebuf, const float* __restrict__ m,
                                  float* __restrict__ z, float* __restrict__ out) {
  int t = blockIdx.x * blockDim.x + threadIdx.x;
  int eidx = t >> 4;
  int sub = t & 15;
  if (eidx >= TE_EDGES_C) return;
  int s, d;
  if (eidx < N_EDGES_C) { s = (int)ei[eidx]; d = (int)ei[(size_t)N_EDGES_C + eidx]; }
  else { s = d = eidx - N_EDGES_C; }
  float a = __expf(ebuf[eidx] - m[d]);
  if (sub == 0) atomicAdd(z + d, a);
  float4 l4 = *(const float4*)(xl + (size_t)s * 64 + sub * 4);
  float* o = out + (size_t)d * 64 + sub * 4;
  atomicAdd(o + 0, a * l4.x);
  atomicAdd(o + 1, a * l4.y);
  atomicAdd(o + 2, a * l4.z);
  atomicAdd(o + 3, a * l4.w);
}

// ---------------- finalize: v = acc/z + bias (+relu); write bf16 (hb) or f32 in place ----------------
__global__ void finalize_kernel(float* __restrict__ acc, const float* __restrict__ z,
                                const float* __restrict__ bias, unsigned short* __restrict__ hb,
                                int do_relu) {
  size_t idx = (size_t)blockIdx.x * blockDim.x + threadIdx.x;
  if (idx >= (size_t)N_NODES_C * 64) return;
  int i = (int)(idx >> 6), c = (int)(idx & 63);
  float v = acc[idx] / z[i] + bias[c];
  if (do_relu) v = v > 0.0f ? v : 0.0f;
  if (hb) hb[idx] = f2bf(v);
  else    acc[idx] = v;
}

// ---------------- global mean pool (sums) ----------------
__global__ void pool_kernel(const float* __restrict__ h, const long long* __restrict__ batch,
                            float* __restrict__ gsum, float* __restrict__ gcnt) {
  size_t idx = (size_t)blockIdx.x * blockDim.x + threadIdx.x;
  if (idx >= (size_t)N_NODES_C * 64) return;
  int i = (int)(idx >> 6), c = (int)(idx & 63);
  int g = (int)batch[i];
  atomicAdd(gsum + (size_t)g * 64 + c, h[idx]);
  if (c == 0) atomicAdd(gcnt + g, 1.0f);
}

// ---------------- head: y = mean(g) @ Wfc1 + b ----------------
__global__ void head1_kernel(const float* __restrict__ gsum, const float* __restrict__ gcnt,
                             const float* __restrict__ W, const float* __restrict__ b,
                             float* __restrict__ y) {
  __shared__ float gm[64];
  int r = blockIdx.x, c = threadIdx.x;
  float cnt = gcnt[r]; if (cnt < 1.0f) cnt = 1.0f;
  gm[c] = gsum[(size_t)r * 64 + c] / cnt;
  __syncthreads();
  float acc = b[c];
#pragma unroll 8
  for (int k = 0; k < 64; ++k) acc += gm[k] * W[k * 64 + c];
  y[(size_t)r * 64 + c] = acc;
}

// batch stats over 512 rows, per channel
__global__ void head2_kernel(const float* __restrict__ y, float* __restrict__ mu,
                             float* __restrict__ rstd) {
  int c = threadIdx.x;
  float s = 0.0f, s2 = 0.0f;
  for (int r = 0; r < NGRAPH_C; ++r) {
    float v = y[(size_t)r * 64 + c];
    s += v; s2 += v * v;
  }
  float m = s / (float)NGRAPH_C;
  float var = s2 / (float)NGRAPH_C - m * m;
  mu[c] = m;
  rstd[c] = rsqrtf(var + 1e-5f);
}

// bn -> relu -> fc2 -> log_softmax
__global__ void head3_kernel(const float* __restrict__ y, const float* __restrict__ gamma,
                             const float* __restrict__ beta, const float* __restrict__ mu,
                             const float* __restrict__ rstd, const float* __restrict__ W2,
                             const float* __restrict__ b2, float* __restrict__ out) {
  int r = blockIdx.x, o = threadIdx.x;   // 16 threads
  float acc = b2[o];
  for (int k = 0; k < 64; ++k) {
    float v = gamma[k] * (y[(size_t)r * 64 + k] - mu[k]) * rstd[k] + beta[k];
    v = v > 0.0f ? v : 0.0f;
    acc += v * W2[k * 16 + o];
  }
  float mx = acc;
#pragma unroll
  for (int off = 8; off >= 1; off >>= 1) mx = fmaxf(mx, __shfl_xor(mx, off, 16));
  float ex = __expf(acc - mx);
  float se = ex;
#pragma unroll
  for (int off = 8; off >= 1; off >>= 1) se += __shfl_xor(se, off, 16);
  out[(size_t)r * 16 + o] = (acc - mx) - __logf(se);
}

// ---------------- host ----------------
extern "C" void kernel_launch(void* const* d_in, const int* in_sizes, int n_in,
                              void* d_out, int out_size, void* d_ws, size_t ws_size,
                              hipStream_t stream) {
  (void)in_sizes; (void)n_in; (void)out_size; (void)ws_size;
  const float*     x     = (const float*)d_in[0];
  const long long* ei    = (const long long*)d_in[1];
  const long long* batch = (const long long*)d_in[2];
  const float *Wl1 = (const float*)d_in[3],  *bl1 = (const float*)d_in[4];
  const float *Wr1 = (const float*)d_in[5],  *br1 = (const float*)d_in[6];
  const float *att1 = (const float*)d_in[7], *bias1 = (const float*)d_in[8];
  const float *Wl2 = (const float*)d_in[9],  *bl2 = (const float*)d_in[10];
  const float *Wr2 = (const float*)d_in[11], *br2 = (const float*)d_in[12];
  const float *att2 = (const float*)d_in[13],*bias2 = (const float*)d_in[14];
  const float *Wfc1 = (const float*)d_in[15],*bfc1 = (const float*)d_in[16];
  const float *gamma = (const float*)d_in[17],*beta = (const float*)d_in[18];
  const float *Wfc2 = (const float*)d_in[19],*bfc2 = (const float*)d_in[20];
  float* out = (float*)d_out;

  // workspace carving (256B aligned); total ~123 MB, L2-resident hot set
  char* ws = (char*)d_ws;
  size_t off = 0;
  auto take = [&](size_t bytes) -> char* {
    char* p = ws + off;
    off = (off + bytes + 255) & ~(size_t)255;
    return p;
  };
  unsigned short* XB  = (unsigned short*)take((size_t)N_NODES_C * 128 * 2);
  unsigned short* HB  = (unsigned short*)take((size_t)N_NODES_C * 64 * 2);
  unsigned short* WLB = (unsigned short*)take((size_t)64 * 128 * 2);
  unsigned short* WRB = (unsigned short*)take((size_t)64 * 128 * 2);
  float* XL   = (float*)take((size_t)N_NODES_C * 64 * 4);
  float* XR   = (float*)take((size_t)N_NODES_C * 64 * 4);
  float* ACC  = (float*)take((size_t)N_NODES_C * 64 * 4);
  float* EB   = (float*)take((size_t)TE_EDGES_C * 4);
  float* M    = (float*)take((size_t)N_NODES_C * 4);
  float* Z    = (float*)take((size_t)N_NODES_C * 4);
  float* GSUM = (float*)take((size_t)NGRAPH_C * 64 * 4);
  float* GCNT = (float*)take((size_t)NGRAPH_C * 4);
  float* Y    = (float*)take((size_t)NGRAPH_C * 64 * 4);
  float* MU   = (float*)take(64 * 4);
  float* RSTD = (float*)take(64 * 4);

  const int gemm_blocks = (N_NODES_C + 31) / 32;
  const int edge_blocks = (TE_EDGES_C + 15) / 16;      // 16 threads per edge
  const int nc_blocks   = ((size_t)N_NODES_C * 64 + 255) / 256;

  // ---- layer 1 ----
  convX_kernel<<<((size_t)N_NODES_C * 128 + 255) / 256, 256, 0, stream>>>(x, XB, (size_t)N_NODES_C * 128);
  convW_kernel<<<(128 * 64 + 255) / 256, 256, 0, stream>>>(Wl1, WLB, 128);
  convW_kernel<<<(128 * 64 + 255) / 256, 256, 0, stream>>>(Wr1, WRB, 128);
  gemm_dual_kernel<128><<<gemm_blocks, 256, 0, stream>>>(XB, WLB, WRB, bl1, br1, XL, XR, N_NODES_C);
  fill_kernel<<<(N_NODES_C + 255) / 256, 256, 0, stream>>>(M, -INFINITY, N_NODES_C);
  fill_kernel<<<(N_NODES_C + 255) / 256, 256, 0, stream>>>(Z, 0.0f, N_NODES_C);
  fill_kernel<<<nc_blocks, 256, 0, stream>>>(ACC, 0.0f, (size_t)N_NODES_C * 64);
  edge_pass1_kernel<<<edge_blocks, 256, 0, stream>>>(ei, XL, XR, att1, EB, M);
  edge_pass2_kernel<<<edge_blocks, 256, 0, stream>>>(ei, XL, EB, M, Z, ACC);
  finalize_kernel<<<nc_blocks, 256, 0, stream>>>(ACC, Z, bias1, HB, 1);   // relu -> bf16 for next gemm

  // ---- layer 2 ----
  convW_kernel<<<(64 * 64 + 255) / 256, 256, 0, stream>>>(Wl2, WLB, 64);
  convW_kernel<<<(64 * 64 + 255) / 256, 256, 0, stream>>>(Wr2, WRB, 64);
  gemm_dual_kernel<64><<<gemm_blocks, 256, 0, stream>>>(HB, WLB, WRB, bl2, br2, XL, XR, N_NODES_C);
  fill_kernel<<<(N_NODES_C + 255) / 256, 256, 0, stream>>>(M, -INFINITY, N_NODES_C);
  fill_kernel<<<(N_NODES_C + 255) / 256, 256, 0, stream>>>(Z, 0.0f, N_NODES_C);
  fill_kernel<<<nc_blocks, 256, 0, stream>>>(ACC, 0.0f, (size_t)N_NODES_C * 64);
  edge_pass1_kernel<<<edge_blocks, 256, 0, stream>>>(ei, XL, XR, att2, EB, M);
  edge_pass2_kernel<<<edge_blocks, 256, 0, stream>>>(ei, XL, EB, M, Z, ACC);
  finalize_kernel<<<nc_blocks, 256, 0, stream>>>(ACC, Z, bias2, (unsigned short*)nullptr, 0); // f32 in place

  // ---- pool + head ----
  fill_kernel<<<(NGRAPH_C * 64 + 255) / 256, 256, 0, stream>>>(GSUM, 0.0f, (size_t)NGRAPH_C * 64);
  fill_kernel<<<(NGRAPH_C + 255) / 256, 256, 0, stream>>>(GCNT, 0.0f, NGRAPH_C);
  pool_kernel<<<nc_blocks, 256, 0, stream>>>(ACC, batch, GSUM, GCNT);
  head1_kernel<<<NGRAPH_C, 64, 0, stream>>>(GSUM, GCNT, Wfc1, bfc1, Y);
  head2_kernel<<<1, 64, 0, stream>>>(Y, MU, RSTD);
  head3_kernel<<<NGRAPH_C, 16, 0, stream>>>(Y, gamma, beta, MU, RSTD, Wfc2, bfc2, out);
}
